// MixBlock_83021717832540
// MI455X (gfx1250) — matrix operand order repaired and checked by reading, not compiled, not run on hardware
//
#include <hip/hip_runtime.h>

// ---------------------------------------------------------------------------
// MixBlock (StyleGAN2 up-block) for MI455X / gfx1250, wave32 + WMMA f16.
//
//   1) style_kernel    : s = istyle @ ws.T + bs            -> styles[2][8][256] f32
//   2) modulate_kernel : wmod[b][o][tap*256+c] f16 (demodulated)
//   3) zero_halo       : clear 1-px halo of padded NHWC images
//   4) upsample_kernel : bilinear x2 (half-pixel), NCHW f32 -> padded NHWC f16
//   5) modconv<false>  : conv1 implicit GEMM (v_wmma_f32_16x16x32_f16),
//                        leaky-ReLU, padded NHWC f16 intermediate
//   6) modconv<true>   : conv2 same, leaky-ReLU, NCHW f32 final output
//
// Padded images give branch-free, fully linear K-loops. 64x64 wave tiles
// (4x4 WMMA accumulators) give 1.0 b128-loads per WMMA from the L2-resident
// operands (weights 9.4 MB, activations 69 MB each -- all inside 192 MB L2).
// ---------------------------------------------------------------------------

typedef __attribute__((ext_vector_type(16))) _Float16 v16h;
typedef __attribute__((ext_vector_type(8)))  _Float16 v8h;
typedef __attribute__((ext_vector_type(8)))  float    v8f;

#define Bn   8
#define LAT  512
#define Cch  256          // input channels (both convs)
#define Och  256          // output channels (both convs)
#define Hh   128          // upsampled height
#define Ww   128          // upsampled width
#define Hp   130          // padded (1-px halo)
#define Wp   130
#define KT   (9 * Cch)    // im2col K = 2304, layout k = tap*C + c
#define LEAKK 0.2f

// ---------------------------------------------------------------------------
// 1) styles[s][b][c] = dot(istyle[b,:], ws[c,:]) + bs[c]
// ---------------------------------------------------------------------------
__global__ __launch_bounds__(256) void style_kernel(
    const float* __restrict__ istyle,
    const float* __restrict__ ws1, const float* __restrict__ bs1,
    const float* __restrict__ ws2, const float* __restrict__ bs2,
    float* __restrict__ styles)
{
    const int b = blockIdx.x;
    const int s = blockIdx.y;
    const int c = threadIdx.x;
    const float* ws = s ? ws2 : ws1;
    const float* bs = s ? bs2 : bs1;
    const float* iv = istyle + (size_t)b * LAT;
    const float* wr = ws + (size_t)c * LAT;
    float acc = 0.f;
#pragma unroll 8
    for (int l = 0; l < LAT; ++l) acc = fmaf(iv[l], wr[l], acc);
    styles[((size_t)s * Bn + b) * Cch + c] = acc + bs[c];
}

// ---------------------------------------------------------------------------
// 2) StyleGAN2 modulate + demodulate -> f16 weights, layout [b][o][tap*C + c]
// ---------------------------------------------------------------------------
__global__ __launch_bounds__(256) void modulate_kernel(
    const float* __restrict__ w,        // [O][C][3][3]
    const float* __restrict__ style_b,  // [B][C]
    _Float16* __restrict__ wmod)        // [B][O][KT] f16
{
    const int o = blockIdx.x;
    const int b = blockIdx.y;
    const int c = threadIdx.x;

    const float sc = style_b[(size_t)b * Cch + c] + 1.0f;
    const float* wp = w + ((size_t)o * Cch + c) * 9;

    float v[9];
    float ss = 0.f;
#pragma unroll
    for (int t = 0; t < 9; ++t) {
        v[t] = wp[t] * sc;
        ss = fmaf(v[t], v[t], ss);
    }

    __shared__ float red[256];
    red[c] = ss;
    __syncthreads();
#pragma unroll
    for (int off = 128; off > 0; off >>= 1) {
        if (c < off) red[c] += red[c + off];
        __syncthreads();
    }
    const float d = rsqrtf(red[0] + 1e-8f);

    _Float16* dst = wmod + ((size_t)b * Och + o) * KT + c;
#pragma unroll
    for (int t = 0; t < 9; ++t)
        dst[(size_t)t * Cch] = (_Float16)(v[t] * d);
}

// ---------------------------------------------------------------------------
// 3) Zero the 1-pixel halo of a padded NHWC image [B][Hp][Wp][C].
// ---------------------------------------------------------------------------
__global__ __launch_bounds__(256) void zero_halo_kernel(_Float16* __restrict__ img)
{
    const int yy = blockIdx.x;
    const int b  = blockIdx.y;
    const int t  = threadIdx.x;
    _Float16* base = img + ((size_t)b * Hp + yy) * Wp * Cch;
    v8h z = {};
    if (yy == 0 || yy == Hp - 1) {
        for (int i = t; i < (Wp * Cch) / 8; i += 256)
            *(v8h*)(base + (size_t)i * 8) = z;
    } else if (t < 64) {
        const int col = t >> 5;                 // 0 or 1
        const int idx = t & 31;
        *(v8h*)(base + (size_t)(col ? (Wp - 1) : 0) * Cch + (size_t)idx * 8) = z;
    }
}

// ---------------------------------------------------------------------------
// 4) Bilinear x2 upsample (align_corners=False), NCHW f32 -> padded NHWC f16.
// ---------------------------------------------------------------------------
__global__ __launch_bounds__(256) void upsample_kernel(
    const float* __restrict__ x,      // [B][C][64][64] f32
    _Float16* __restrict__ xup)       // [B][Hp][Wp][C] f16 (interior)
{
    const int y = blockIdx.x;
    const int b = blockIdx.y;
    const int t = threadIdx.x;

    const int j = y >> 1;
    int ya, yb; float wyA, wyB;
    if ((y & 1) == 0) { ya = j - 1; yb = j;     wyA = 0.25f; wyB = 0.75f; }
    else              { ya = j;     yb = j + 1; wyA = 0.75f; wyB = 0.25f; }
    ya = ya < 0 ? 0 : ya;
    yb = yb > 63 ? 63 : yb;

    __shared__ float stage[16 * 2 * 64];   // [c_local(16)][row(2)][x(64)]

    const int xo  = t & 127;
    const int sub = t >> 7;
    const int jx  = xo >> 1;
    int xa, xb; float wxA, wxB;
    if ((xo & 1) == 0) { xa = jx - 1; xb = jx;     wxA = 0.25f; wxB = 0.75f; }
    else               { xa = jx;     xb = jx + 1; wxA = 0.75f; wxB = 0.25f; }
    xa = xa < 0 ? 0 : xa;
    xb = xb > 63 ? 63 : xb;

    _Float16* orow = xup + (((size_t)b * Hp + (y + 1)) * Wp + (xo + 1)) * Cch;

    for (int cc = 0; cc < Cch; cc += 16) {
#pragma unroll
        for (int i = 0; i < 8; ++i) {
            const int idx = t * 8 + i;
            const int cl  = idx >> 7;
            const int rem = idx & 127;
            const int row = rem >> 6;
            const int xi  = rem & 63;
            const int ysrc = row == 0 ? ya : yb;
            stage[idx] = x[(((size_t)b * Cch + cc + cl) * 64 + ysrc) * 64 + xi];
        }
        __syncthreads();

        v8h hv;
#pragma unroll
        for (int k = 0; k < 8; ++k) {
            const int cl = sub * 8 + k;
            const float* s0 = &stage[(cl * 2 + 0) * 64];
            const float* s1 = &stage[(cl * 2 + 1) * 64];
            const float v = wyA * (wxA * s0[xa] + wxB * s0[xb]) +
                            wyB * (wxA * s1[xa] + wxB * s1[xb]);
            hv[k] = (_Float16)v;
        }
        *(v8h*)(orow + cc + sub * 8) = hv;
        __syncthreads();
    }
}

// ---------------------------------------------------------------------------
// 5/6) Modulated 3x3 conv as implicit GEMM with v_wmma_f32_16x16x32_f16.
//   Branch-free (zero halo = conv padding). Block: 256 thr = 8 waves
//   (4 M-waves x 2 N-waves), wave tile 64x64 -> 4x4 WMMA accumulators.
//   Block tile: all 256 output channels x 128 pixels (one row).
//   Grid: (128 rows, 8 batches). K: 9 taps x 8 chunks of 32 (fully linear).
//   Fragment layouts used here (A and B share one k order):
//     A lane(m=l%16, half=l/16): f16 e -> K = e + 16*half  (contiguous)
//     B lane(n=l%16, half):      f16 e -> K = e + 16*half  (contiguous)
//     C/D lane: N = l%16, VGPR r -> M = r + 8*half
// ---------------------------------------------------------------------------
template <bool FP32_OUT>
__global__ __launch_bounds__(256) void modconv_kernel(
    const _Float16* __restrict__ xin,   // [B][Hp][Wp][C] f16, zero halo
    const _Float16* __restrict__ wmod,  // [B][O][KT]     f16
    _Float16* __restrict__ out_h,       // [B][Hp][Wp][O] f16 (if !FP32_OUT)
    float* __restrict__ out_f)          // [B][O][H][W]   f32 (if FP32_OUT)
{
    const int h = blockIdx.x;             // output row (unpadded coords)
    const int b = blockIdx.y;

    const int tid  = threadIdx.x;
    const int wave = tid >> 5;
    const int lane = tid & 31;
    const int lm   = lane & 15;
    const int half = lane >> 4;
    const int wm   = wave & 3;            // M wave 0..3 : 64 rows each
    const int wn   = wave >> 2;           // N wave 0..1 : 64 px each
    const int xbase = wn * 64;

    v8f zero8 = {};
    v8f acc[4][4];
#pragma unroll
    for (int mi = 0; mi < 4; ++mi)
#pragma unroll
        for (int ni = 0; ni < 4; ++ni) acc[mi][ni] = zero8;

    // A base: row o = wm*64 + mi*16 + lm; mi offsets are immediates (16*KT elems).
    const _Float16* pA = wmod +
        ((size_t)b * Och + (wm * 64 + lm)) * KT + half * 16;

    const _Float16* xb_img = xin + (size_t)b * Hp * Wp * Cch + half * 16;

#pragma unroll 1
    for (int tap = 0; tap < 9; ++tap) {
        const int dy = tap / 3;           // padded row = h + dy  (0..129)
        const int dx = tap % 3;           // padded col = x + dx  (0..129)
        const _Float16* pB = xb_img +
            ((size_t)(h + dy) * Wp + (xbase + lm + dx)) * Cch;

#pragma unroll 2
        for (int cc8 = 0; cc8 < 8; ++cc8) {
            v16h A[4];
#pragma unroll
            for (int mi = 0; mi < 4; ++mi) {
                const _Float16* p = pA + (size_t)mi * 16 * KT;
                v8h lo = *(const v8h*)(p);        // K {0..7}  + 16*half
                v8h hi = *(const v8h*)(p + 8);    // K {8..15} + 16*half
                A[mi] = __builtin_shufflevector(lo, hi,
                        0, 1, 2, 3, 4, 5, 6, 7, 8, 9, 10, 11, 12, 13, 14, 15);
            }

#pragma unroll
            for (int ni = 0; ni < 4; ++ni) {
                v16h Bf = *(const v16h*)(pB + (size_t)ni * 16 * Cch);
#pragma unroll
                for (int mi = 0; mi < 4; ++mi) {
                    acc[mi][ni] = __builtin_amdgcn_wmma_f32_16x16x32_f16(
                        false, A[mi], false, Bf, (short)0, acc[mi][ni],
                        false, false);
                }
            }
            pA += 32; pB += 32;
        }
    }

    // Epilogue: leaky-ReLU + store.
#pragma unroll
    for (int mi = 0; mi < 4; ++mi) {
        const int o0 = wm * 64 + mi * 16 + half * 8;
#pragma unroll
        for (int ni = 0; ni < 4; ++ni) {
            const int x = xbase + ni * 16 + lm;
            if constexpr (FP32_OUT) {
#pragma unroll
                for (int r = 0; r < 8; ++r) {
                    float v = acc[mi][ni][r];
                    v = v > 0.f ? v : LEAKK * v;
                    out_f[(((size_t)b * Och + (o0 + r)) * Hh + h) * Ww + x] = v;
                }
            } else {
                v8h hv;
#pragma unroll
                for (int r = 0; r < 8; ++r) {
                    float v = acc[mi][ni][r];
                    v = v > 0.f ? v : LEAKK * v;
                    hv[r] = (_Float16)v;
                }
                *(v8h*)(out_h + (((size_t)b * Hp + (h + 1)) * Wp + (x + 1)) * Cch + o0) = hv;
            }
        }
    }
}

// ---------------------------------------------------------------------------
extern "C" void kernel_launch(void* const* d_in, const int* in_sizes, int n_in,
                              void* d_out, int out_size, void* d_ws, size_t ws_size,
                              hipStream_t stream) {
    const float* x      = (const float*)d_in[0];   // [8][256][64][64]
    const float* istyle = (const float*)d_in[1];   // [8][512]
    const float* ws1    = (const float*)d_in[2];   // [256][512]
    const float* bs1    = (const float*)d_in[3];   // [256]
    const float* w1     = (const float*)d_in[4];   // [256][256][3][3]
    const float* ws2    = (const float*)d_in[5];
    const float* bs2    = (const float*)d_in[6];
    const float* w2     = (const float*)d_in[7];
    float* out = (float*)d_out;                    // [8][256][128][128]

    char* ws = (char*)d_ws;
    const size_t styles_bytes = (size_t)2 * Bn * Cch * sizeof(float);       // 16 KB
    const size_t wmod_bytes   = (size_t)Bn * Och * KT * sizeof(_Float16);   // 9.4 MB
    const size_t img_bytes    = (size_t)Bn * Hp * Wp * Cch * sizeof(_Float16); // 69.2 MB

    float*    styles = (float*)(ws);
    _Float16* wmod1  = (_Float16*)(ws + styles_bytes);
    _Float16* wmod2  = (_Float16*)(ws + styles_bytes + wmod_bytes);
    _Float16* xup    = (_Float16*)(ws + styles_bytes + 2 * wmod_bytes);
    _Float16* xmid   = (_Float16*)(ws + styles_bytes + 2 * wmod_bytes + img_bytes);

    style_kernel<<<dim3(Bn, 2), 256, 0, stream>>>(istyle, ws1, bs1, ws2, bs2, styles);
    modulate_kernel<<<dim3(Och, Bn), 256, 0, stream>>>(w1, styles,            wmod1);
    modulate_kernel<<<dim3(Och, Bn), 256, 0, stream>>>(w2, styles + Bn * Cch, wmod2);
    zero_halo_kernel<<<dim3(Hp, Bn), 256, 0, stream>>>(xup);
    zero_halo_kernel<<<dim3(Hp, Bn), 256, 0, stream>>>(xmid);
    upsample_kernel<<<dim3(Hh, Bn), 256, 0, stream>>>(x, xup);

    modconv_kernel<false><<<dim3(Hh, Bn), 256, 0, stream>>>(xup,  wmod1, xmid, nullptr);
    modconv_kernel<true ><<<dim3(Hh, Bn), 256, 0, stream>>>(xmid, wmod2, nullptr, out);
}
